// SBlock1_61598420959843
// MI455X (gfx1250) — matrix operand order repaired and checked by reading, not compiled, not run on hardware
//
#include <hip/hip_runtime.h>
#include <hip/hip_bf16.h>

typedef __attribute__((ext_vector_type(16))) _Float16 v16h;
typedef __attribute__((ext_vector_type(8)))  float    v8f;

#define B_  64
#define S_  1024
#define F_  768
#define H_  12
#define DH_ 64
#define P2_ 64

// ---------------- WMMA fragment loaders (16x16x32 f16) ----------------
// A (16x32 MxK): lanes 0-15 row M=l, K={k0..k0+7, k0+16..k0+23};
//                lanes 16-31 row M=l, K={k0+8..k0+15, k0+24..k0+31}.
__device__ __forceinline__ v16h a_frag_from_f32(const float* row, int k0, int hi) {
  v16h a;
  int base = k0 + hi * 8;
#pragma unroll
  for (int i = 0; i < 8; ++i) {
    a[i]     = (_Float16)row[base + i];
    a[i + 8] = (_Float16)row[base + 16 + i];
  }
  return a;
}
__device__ __forceinline__ v16h a_frag_from_f16(const _Float16* row, int k0, int hi) {
  v16h a;
  int base = k0 + hi * 8;
#pragma unroll
  for (int i = 0; i < 8; ++i) {
    a[i]     = row[base + i];
    a[i + 8] = row[base + 16 + i];
  }
  return a;
}
// B (32x16 KxN): lane l holds column N=n0+l; lanes 0-15 K={k0..k0+7, k0+16..k0+23},
//                lanes 16-31 K={k0+8..k0+15, k0+24..k0+31}. btrow points at Bt[n] (N x K row-major).
__device__ __forceinline__ v16h b_frag_from_f32(const float* btrow, int k0, int hi) {
  v16h b;
  int base = k0 + hi * 8;
#pragma unroll
  for (int i = 0; i < 8; ++i) {
    b[i]     = (_Float16)btrow[base + i];
    b[i + 8] = (_Float16)btrow[base + 16 + i];
  }
  return b;
}
__device__ __forceinline__ v16h b_frag_from_f16(const _Float16* btrow, int k0, int hi) {
  v16h b;
  int base = k0 + hi * 8;
#pragma unroll
  for (int i = 0; i < 8; ++i) {
    b[i]     = btrow[base + i];
    b[i + 8] = btrow[base + 16 + i];
  }
  return b;
}

// ---------------- 1) tail copy: out[:,64:,:] = x[:,64:,:] ----------------
__global__ void __launch_bounds__(256) copy_tail_kernel(const float* __restrict__ x,
                                                        float* __restrict__ out) {
  const int PER_B4  = (S_ - P2_) * F_ / 4;                 // 184320 float4 per batch
  const long long TOTAL4 = (long long)B_ * PER_B4;         // 11796480
  long long i = (long long)blockIdx.x * blockDim.x + threadIdx.x;
  if (i >= TOTAL4) return;
  int b   = (int)(i / PER_B4);
  int rem = (int)(i % PER_B4);
  long long off = (long long)b * (S_ * F_ / 4) + (P2_ * F_ / 4) + rem;
  ((float4*)out)[off] = ((const float4*)x)[off];
}

// ---------------- 2) QKV projection: one 32x32 tile per wave ----------------
// 4 WMMAs per K-step (2 A-frags x 2 B-frags) => 2x arithmetic intensity vs 16x16.
__global__ void __launch_bounds__(256) qkv_kernel(const float* __restrict__ x,
    const float* __restrict__ Wq, const float* __restrict__ bq,
    const float* __restrict__ Wk, const float* __restrict__ bk,
    const float* __restrict__ Wv, const float* __restrict__ bv,
    _Float16* __restrict__ qws, _Float16* __restrict__ kws, _Float16* __restrict__ vws) {
  const int wave = threadIdx.x >> 5;
  const int lane = threadIdx.x & 31;
  const int l = lane & 15, hi = lane >> 4;
  const int MT = (B_ * P2_) / 32, NT = F_ / 32;            // 128, 24
  int tile = blockIdx.x * 8 + wave;                        // 9216 total (exact)
  int mat  = tile / (MT * NT);
  int rest = tile % (MT * NT);
  int mt = rest / NT, nt = rest % NT;

  const float* W    = (mat == 0) ? Wq : (mat == 1) ? Wk : Wv;
  const float* bias = (mat == 0) ? bq : (mat == 1) ? bk : bv;
  const float scale = (mat == 0) ? 0.125f : 1.0f;          // fold 1/SCALE into Q

  const int m0 = mt * 32, n0 = nt * 32;
  // A rows for the two M sub-tiles (x rows: b = m>>6, s = m&63)
  int ma0 = m0 + l,      ba0 = ma0 >> 6, sa0 = ma0 & 63;
  int ma1 = m0 + 16 + l, ba1 = ma1 >> 6, sa1 = ma1 & 63;
  const float* arow0 = x + ((long long)ba0 * S_ + sa0) * F_;
  const float* arow1 = x + ((long long)ba1 * S_ + sa1) * F_;
  // B rows for the two N sub-tiles (W[h,d,:] rows, N x K row-major)
  const float* brow0 = W + (long long)(n0 + l) * F_;
  const float* brow1 = W + (long long)(n0 + 16 + l) * F_;

  v8f acc00 = {}, acc01 = {}, acc10 = {}, acc11 = {};
#pragma unroll 2
  for (int k0 = 0; k0 < F_; k0 += 32) {
    v16h a0 = a_frag_from_f32(arow0, k0, hi);
    v16h a1 = a_frag_from_f32(arow1, k0, hi);
    v16h b0 = b_frag_from_f32(brow0, k0, hi);
    v16h b1 = b_frag_from_f32(brow1, k0, hi);
    acc00 = __builtin_amdgcn_wmma_f32_16x16x32_f16(false, a0, false, b0, (short)0, acc00, false, false);
    acc01 = __builtin_amdgcn_wmma_f32_16x16x32_f16(false, a0, false, b1, (short)0, acc01, false, false);
    acc10 = __builtin_amdgcn_wmma_f32_16x16x32_f16(false, a1, false, b0, (short)0, acc10, false, false);
    acc11 = __builtin_amdgcn_wmma_f32_16x16x32_f16(false, a1, false, b1, (short)0, acc11, false, false);
  }

  v8f accs[2][2] = {{acc00, acc01}, {acc10, acc11}};
#pragma unroll
  for (int i = 0; i < 2; ++i) {
#pragma unroll
    for (int j = 0; j < 2; ++j) {
      int n = n0 + j * 16 + l;
      int h = n >> 6, d = n & 63;
      float bval = bias[n];
#pragma unroll
      for (int r = 0; r < 8; ++r) {
        int m = m0 + i * 16 + r + hi * 8;                  // C: lanes 16-31 -> M=r+8
        int bb = m >> 6, ss = m & 63;
        _Float16 hv = (_Float16)((accs[i][j][r] + bval) * scale);
        if (mat == 2)      vws[(((long long)bb * H_ + h) * DH_ + d) * P2_ + ss] = hv; // V: [b,h,d,t]
        else if (mat == 1) kws[(((long long)bb * H_ + h) * P2_ + ss) * DH_ + d] = hv; // K: [b,h,s,d]
        else               qws[(((long long)bb * H_ + h) * P2_ + ss) * DH_ + d] = hv; // Q: [b,h,s,d]
      }
    }
  }
}

// ---------------- 3) attention per (b,h): 4 waves / block ----------------
__global__ void __launch_bounds__(128) attn_kernel(const _Float16* __restrict__ qws,
                                                   const _Float16* __restrict__ kws,
                                                   const _Float16* __restrict__ vws,
                                                   _Float16* __restrict__ ows) {
  __shared__ float    sc[P2_ * P2_];      // 16 KB scores
  __shared__ _Float16 at[P2_ * P2_];      //  8 KB attn weights (f16)
  int bh = blockIdx.x;                    // 0..767
  int bb = bh / H_, h = bh % H_;
  const _Float16* qp = qws + (long long)bh * P2_ * DH_;
  const _Float16* kp = kws + (long long)bh * P2_ * DH_;
  const _Float16* vp = vws + (long long)bh * DH_ * P2_;

  int wave = threadIdx.x >> 5;
  int lane = threadIdx.x & 31;
  int l = lane & 15, hi = lane >> 4;
  int row0 = wave * 16;

  // scores = (Q/8) * K^T  -> LDS
  const _Float16* arow = qp + (row0 + l) * DH_;
#pragma unroll
  for (int ct = 0; ct < 4; ++ct) {
    const _Float16* brow = kp + (ct * 16 + l) * DH_;       // K[t,:] row = Bt[n=t]
    v8f acc = {};
#pragma unroll
    for (int k0 = 0; k0 < DH_; k0 += 32) {
      v16h a = a_frag_from_f16(arow, k0, hi);
      v16h b = b_frag_from_f16(brow, k0, hi);
      acc = __builtin_amdgcn_wmma_f32_16x16x32_f16(false, a, false, b, (short)0, acc, false, false);
    }
#pragma unroll
    for (int r = 0; r < 8; ++r)
      sc[(row0 + r + hi * 8) * P2_ + ct * 16 + l] = acc[r];
  }
  __syncthreads();

  // row softmax (64 rows handled by threads 0..63)
  if (threadIdx.x < P2_) {
    int r = threadIdx.x;
    float mx = -1e30f;
    for (int c = 0; c < P2_; ++c) mx = fmaxf(mx, sc[r * P2_ + c]);
    float sum = 0.f;
    for (int c = 0; c < P2_; ++c) { float e = expf(sc[r * P2_ + c] - mx); sc[r * P2_ + c] = e; sum += e; }
    float inv = 1.f / sum;
    for (int c = 0; c < P2_; ++c) at[r * P2_ + c] = (_Float16)(sc[r * P2_ + c] * inv);
  }
  __syncthreads();

  // O = attn * V ; stage as [b, s, h*DH + d] f16 for projection
  const _Float16* arow2 = at + (row0 + l) * P2_;
#pragma unroll
  for (int ct = 0; ct < 4; ++ct) {
    const _Float16* brow = vp + (ct * 16 + l) * P2_;       // V[d,:] over t = Bt[n=d]
    v8f acc = {};
#pragma unroll
    for (int k0 = 0; k0 < P2_; k0 += 32) {
      v16h a = a_frag_from_f16(arow2, k0, hi);
      v16h b = b_frag_from_f16(brow, k0, hi);
      acc = __builtin_amdgcn_wmma_f32_16x16x32_f16(false, a, false, b, (short)0, acc, false, false);
    }
#pragma unroll
    for (int r = 0; r < 8; ++r) {
      int s = row0 + r + hi * 8;
      int d = ct * 16 + l;
      ows[((long long)bb * P2_ + s) * F_ + h * DH_ + d] = (_Float16)acc[r];
    }
  }
}

// ---------------- 4) output projection: O @ Wo^T + bo, 32x32 per wave ----------------
__global__ void __launch_bounds__(256) proj_kernel(const _Float16* __restrict__ ows,
                                                   const float* __restrict__ Wo,
                                                   const float* __restrict__ bo,
                                                   float* __restrict__ out) {
  const int wave = threadIdx.x >> 5;
  const int lane = threadIdx.x & 31;
  const int l = lane & 15, hi = lane >> 4;
  const int NT = F_ / 32;                                  // 24
  int tile = blockIdx.x * 8 + wave;                        // 3072 total (exact)
  int mt = tile / NT, nt = tile % NT;
  const int m0 = mt * 32, n0 = nt * 32;

  const _Float16* arow0 = ows + (long long)(m0 + l) * F_;
  const _Float16* arow1 = ows + (long long)(m0 + 16 + l) * F_;
  const float*    brow0 = Wo  + (long long)(n0 + l) * F_;        // Wo[n,:] row (N x K)
  const float*    brow1 = Wo  + (long long)(n0 + 16 + l) * F_;

  v8f acc00 = {}, acc01 = {}, acc10 = {}, acc11 = {};
#pragma unroll 2
  for (int k0 = 0; k0 < F_; k0 += 32) {
    v16h a0 = a_frag_from_f16(arow0, k0, hi);
    v16h a1 = a_frag_from_f16(arow1, k0, hi);
    v16h b0 = b_frag_from_f32(brow0, k0, hi);
    v16h b1 = b_frag_from_f32(brow1, k0, hi);
    acc00 = __builtin_amdgcn_wmma_f32_16x16x32_f16(false, a0, false, b0, (short)0, acc00, false, false);
    acc01 = __builtin_amdgcn_wmma_f32_16x16x32_f16(false, a0, false, b1, (short)0, acc01, false, false);
    acc10 = __builtin_amdgcn_wmma_f32_16x16x32_f16(false, a1, false, b0, (short)0, acc10, false, false);
    acc11 = __builtin_amdgcn_wmma_f32_16x16x32_f16(false, a1, false, b1, (short)0, acc11, false, false);
  }

  v8f accs[2][2] = {{acc00, acc01}, {acc10, acc11}};
#pragma unroll
  for (int i = 0; i < 2; ++i) {
#pragma unroll
    for (int j = 0; j < 2; ++j) {
      int n = n0 + j * 16 + l;
      float bias = bo[n];
#pragma unroll
      for (int r = 0; r < 8; ++r) {
        int m = m0 + i * 16 + r + hi * 8;
        int bb = m >> 6, ss = m & 63;
        out[((long long)bb * S_ + ss) * F_ + n] = accs[i][j][r] + bias;
      }
    }
  }
}

extern "C" void kernel_launch(void* const* d_in, const int* in_sizes, int n_in,
                              void* d_out, int out_size, void* d_ws, size_t ws_size,
                              hipStream_t stream) {
  const float* x  = (const float*)d_in[0];
  const float* Wq = (const float*)d_in[1];
  const float* bq = (const float*)d_in[2];
  const float* Wk = (const float*)d_in[3];
  const float* bk = (const float*)d_in[4];
  const float* Wv = (const float*)d_in[5];
  const float* bv = (const float*)d_in[6];
  const float* Wo = (const float*)d_in[7];
  const float* bo = (const float*)d_in[8];
  float* out = (float*)d_out;

  const size_t NQ = (size_t)B_ * H_ * P2_ * DH_;           // 3,145,728 halves each
  _Float16* qws = (_Float16*)d_ws;
  _Float16* kws = qws + NQ;
  _Float16* vws = kws + NQ;
  _Float16* ows = vws + NQ;                                // B*P2*F, same count (~25 MB total)

  copy_tail_kernel<<<46080, 256, 0, stream>>>(x, out);     // 11,796,480 float4 (exact)
  qkv_kernel<<<1152, 256, 0, stream>>>(x, Wq, bq, Wk, bk, Wv, bv, qws, kws, vws);  // 9216 waves
  attn_kernel<<<768, 128, 0, stream>>>(qws, kws, vws, ows);
  proj_kernel<<<384, 256, 0, stream>>>(ows, Wo, bo, out);  // 3072 waves
}
